// CLoss_68521908241007
// MI455X (gfx1250) — matrix operand order repaired and checked by reading, not compiled, not run on hardware
//
#include <hip/hip_runtime.h>
#include <hip/hip_bf16.h>
#include <math.h>

// ---------------------------------------------------------------------------
// Fused  L2-distance softmax cross-entropy  for MI455X (gfx1250, wave32).
//
//   loss = mean_i [ log sum_j exp(-t*||x_i - y_j||)  +  t*||x_i - y_lab(i)|| ]
//
// Strategy:
//   1) prep:   split f32 -> bf16 hi/lo (error-compensated), exact row norms.
//   2) picked: exact f32 distance to the labeled column per row.
//   3) main:   WMMA bf16 16x16x32 GEMM tiles (hi*hi + hi*lo + lo*hi) with
//              THREE independent accumulator chains for WMMA ILP, fused with
//              sq = |x|^2 + |y|^2 - 2 x.y -> sqrt -> exp -> row Sigma,
//              never materializing the 8192x8192 logits (saves ~512MB HBM rt).
//   4) reduce: deterministic fixed-order combine + tree reduction -> scalar.
// ---------------------------------------------------------------------------

typedef __bf16 bf16;
typedef __attribute__((ext_vector_type(16))) __bf16 v16bf;
typedef __attribute__((ext_vector_type(8)))  __bf16 v8bf;
typedef __attribute__((ext_vector_type(4)))  __bf16 v4bf;
typedef __attribute__((ext_vector_type(8)))  float  v8f;

#define N_ROWS 8192
#define M_COLS 8192
#define DDIM   128
#define NCHUNK 4
#define COLS_PER_CHUNK (M_COLS / NCHUNK)

__device__ __forceinline__ bf16 f2bf(float f) {   // f32 -> bf16, round-nearest-even
  unsigned u = __float_as_uint(f);
  u += 0x7FFFu + ((u >> 16) & 1u);
  unsigned short s = (unsigned short)(u >> 16);
  return __builtin_bit_cast(bf16, s);
}
__device__ __forceinline__ float bf2f(bf16 b) {
  unsigned short s = __builtin_bit_cast(unsigned short, b);
  return __uint_as_float(((unsigned)s) << 16);
}

// ---------------------------------------------------------------------------
// Kernel 1: bf16 hi/lo split + exact row sum-of-squares. One wave per row.
// Rows [0,8192) = feat, rows [8192,16384) = feat2.
// ---------------------------------------------------------------------------
__global__ void CLoss_prep_kernel(const float* __restrict__ feat,
                                  const float* __restrict__ feat2,
                                  bf16* __restrict__ xh, bf16* __restrict__ xl,
                                  bf16* __restrict__ yh, bf16* __restrict__ yl,
                                  float* __restrict__ xsq, float* __restrict__ ysq) {
  const int wid  = (blockIdx.x * blockDim.x + threadIdx.x) >> 5;
  const int lane = threadIdx.x & 31;
  const float* src; bf16* ph; bf16* pl; float* sq; int row;
  if (wid < N_ROWS) { row = wid;           src = feat;  ph = xh; pl = xl; sq = xsq; }
  else              { row = wid - N_ROWS;  src = feat2; ph = yh; pl = yl; sq = ysq; }

  const float4 v = *(const float4*)(src + (size_t)row * DDIM + lane * 4);
  float ss = v.x*v.x + v.y*v.y + v.z*v.z + v.w*v.w;
  ss += __shfl_xor(ss, 1, 32);
  ss += __shfl_xor(ss, 2, 32);
  ss += __shfl_xor(ss, 4, 32);
  ss += __shfl_xor(ss, 8, 32);
  ss += __shfl_xor(ss, 16, 32);
  if (lane == 0) sq[row] = ss;

  v4bf h, l;
  h[0] = f2bf(v.x); l[0] = f2bf(v.x - bf2f(h[0]));
  h[1] = f2bf(v.y); l[1] = f2bf(v.y - bf2f(h[1]));
  h[2] = f2bf(v.z); l[2] = f2bf(v.z - bf2f(h[2]));
  h[3] = f2bf(v.w); l[3] = f2bf(v.w - bf2f(h[3]));
  *(v4bf*)(ph + (size_t)row * DDIM + lane * 4) = h;
  *(v4bf*)(pl + (size_t)row * DDIM + lane * 4) = l;
}

// ---------------------------------------------------------------------------
// Kernel 2: exact f32 picked logit per row:  picked[i] = -t * ||x_i - y_lab||.
// One wave per row.
// ---------------------------------------------------------------------------
__global__ void CLoss_picked_kernel(const float* __restrict__ feat,
                                    const float* __restrict__ feat2,
                                    const int*   __restrict__ labels,
                                    const float* __restrict__ temp,
                                    float* __restrict__ picked) {
  const int wid  = (blockIdx.x * blockDim.x + threadIdx.x) >> 5;
  const int lane = threadIdx.x & 31;
  const int j = labels[wid];
  const float4 x = *(const float4*)(feat  + (size_t)wid * DDIM + lane * 4);
  const float4 y = *(const float4*)(feat2 + (size_t)j   * DDIM + lane * 4);
  const float dx = x.x - y.x, dy = x.y - y.y, dz = x.z - y.z, dw = x.w - y.w;
  float ss = dx*dx + dy*dy + dz*dz + dw*dw;
  ss += __shfl_xor(ss, 1, 32);
  ss += __shfl_xor(ss, 2, 32);
  ss += __shfl_xor(ss, 4, 32);
  ss += __shfl_xor(ss, 8, 32);
  ss += __shfl_xor(ss, 16, 32);
  if (lane == 0) picked[wid] = -temp[0] * sqrtf(ss);
}

// ---------------------------------------------------------------------------
// Kernel 3: fused WMMA GEMM + distance + exp + per-row partial Sigma.
// Wave = one 16-row tile x one 2048-column chunk. 2048 waves total.
//
// A frag (16x32 bf16): lane&15 = row M, elements e=0..15 -> K = 16*(lane>>4)+e.
// B frag (32x16 bf16): lane&15 = col N, elements e=0..15 -> K = 16*(lane>>4)+e.
// C/D  (16x16 f32):    element v -> row v + 8*(lane>>4), col = lane&15.
// ---------------------------------------------------------------------------
__global__ void __launch_bounds__(256)
CLoss_main_kernel(const bf16* __restrict__ xh, const bf16* __restrict__ xl,
                  const bf16* __restrict__ yh, const bf16* __restrict__ yl,
                  const float* __restrict__ xsq, const float* __restrict__ ysq,
                  const float* __restrict__ temp,
                  float* __restrict__ partial) {
  const int wid   = blockIdx.x * 8 + (threadIdx.x >> 5);
  const int lane  = threadIdx.x & 31;
  const int half  = lane >> 4;
  const int l15   = lane & 15;
  const int tile  = wid >> 2;          // 0..511 : 16-row tile
  const int chunk = wid & 3;           // 0..3   : column chunk
  const int r0    = tile * 16;
  const int c0    = chunk * COLS_PER_CHUNK;
  const float tmp = temp[0];

  // --- A fragments for this row tile: 4 K-chunks of 32, hi + lo. (64 VGPRs)
  v16bf ahi[4], alo[4];
  {
    const int kb = half * 16;
    const bf16* Ah = xh + (size_t)(r0 + l15) * DDIM + kb;
    const bf16* Al = xl + (size_t)(r0 + l15) * DDIM + kb;
#pragma unroll
    for (int kc = 0; kc < 4; ++kc) {
      v8bf h0 = *(const v8bf*)(Ah + kc * 32);
      v8bf h1 = *(const v8bf*)(Ah + kc * 32 + 8);
      v8bf l0 = *(const v8bf*)(Al + kc * 32);
      v8bf l1 = *(const v8bf*)(Al + kc * 32 + 8);
#pragma unroll
      for (int e = 0; e < 8; ++e) {
        ahi[kc][e] = h0[e]; ahi[kc][e + 8] = h1[e];
        alo[kc][e] = l0[e]; alo[kc][e + 8] = l1[e];
      }
    }
  }

  float xsqm[8];
#pragma unroll
  for (int v = 0; v < 8; ++v) xsqm[v] = xsq[r0 + half * 8 + v];

  float se[8];
#pragma unroll
  for (int v = 0; v < 8; ++v) se[v] = 0.0f;

  const bf16* Bh0 = yh + (size_t)l15 * DDIM + half * 16;
  const bf16* Bl0 = yl + (size_t)l15 * DDIM + half * 16;

  for (int j0 = c0; j0 < c0 + COLS_PER_CHUNK; j0 += 16) {
    const bf16* Bh = Bh0 + (size_t)j0 * DDIM;
    const bf16* Bl = Bl0 + (size_t)j0 * DDIM;
    __builtin_prefetch(Bh + 16 * DDIM, 0, 3);   // prefetch next B tile, near scope
    __builtin_prefetch(Bl + 16 * DDIM, 0, 3);

    // Three independent WMMA accumulator chains (depth 4 each) for ILP:
    //   c_hh = Sum xh.yh   c_hl = Sum xh.yl   c_lh = Sum xl.yh
    v8f c_hh = {}, c_hl = {}, c_lh = {};
#pragma unroll
    for (int kc = 0; kc < 4; ++kc) {
      v16bf bh = *(const v16bf*)(Bh + kc * 32);
      v16bf bl = *(const v16bf*)(Bl + kc * 32);
      c_hh = __builtin_amdgcn_wmma_f32_16x16x32_bf16(false, ahi[kc], false, bh,
                                                     (short)0, c_hh, false, false);
      c_hl = __builtin_amdgcn_wmma_f32_16x16x32_bf16(false, ahi[kc], false, bl,
                                                     (short)0, c_hl, false, false);
      c_lh = __builtin_amdgcn_wmma_f32_16x16x32_bf16(false, alo[kc], false, bh,
                                                     (short)0, c_lh, false, false);
    }

    const float ysqj = ysq[j0 + l15];
#pragma unroll
    for (int v = 0; v < 8; ++v) {
      // small terms first for accuracy: dot = (hl + lh) + hh
      float dot = (c_hl[v] + c_lh[v]) + c_hh[v];
      float sq  = xsqm[v] + ysqj - 2.0f * dot;
      float d   = __builtin_amdgcn_sqrtf(fmaxf(sq, 0.0f)); // bare v_sqrt_f32
      se[v] += __expf(-tmp * d);    // logits <= 0 -> shift-0 sum-exp is safe
    }
  }

  // Row totals: butterfly over the 16 lanes that share a row.
#pragma unroll
  for (int v = 0; v < 8; ++v) {
    float s = se[v];
    s += __shfl_xor(s, 1, 32);
    s += __shfl_xor(s, 2, 32);
    s += __shfl_xor(s, 4, 32);
    s += __shfl_xor(s, 8, 32);
    se[v] = s;
  }
  if (l15 == 0) {
#pragma unroll
    for (int v = 0; v < 8; ++v)
      partial[(size_t)chunk * N_ROWS + r0 + half * 8 + v] = se[v];
  }
}

// ---------------------------------------------------------------------------
// Kernel 4: deterministic final reduction (fixed order, single block).
// loss = mean_i ( log(sum_c partial[c][i]) - picked[i] )
// ---------------------------------------------------------------------------
__global__ void CLoss_reduce_kernel(const float* __restrict__ partial,
                                    const float* __restrict__ picked,
                                    float* __restrict__ out) {
  __shared__ float sh[256];
  float a = 0.0f;
  for (int r = threadIdx.x; r < N_ROWS; r += 256) {
    float s = partial[r]
            + partial[1 * N_ROWS + r]
            + partial[2 * N_ROWS + r]
            + partial[3 * N_ROWS + r];
    a += __logf(s) - picked[r];
  }
  sh[threadIdx.x] = a;
  __syncthreads();
  for (int st = 128; st > 0; st >>= 1) {
    if ((int)threadIdx.x < st) sh[threadIdx.x] += sh[threadIdx.x + st];
    __syncthreads();
  }
  if (threadIdx.x == 0) out[0] = sh[0] / (float)N_ROWS;
}

// ---------------------------------------------------------------------------
extern "C" void kernel_launch(void* const* d_in, const int* in_sizes, int n_in,
                              void* d_out, int out_size, void* d_ws, size_t ws_size,
                              hipStream_t stream) {
  const float* feat   = (const float*)d_in[0];
  const float* feat2  = (const float*)d_in[1];
  const float* temp   = (const float*)d_in[2];
  const int*   labels = (const int*)d_in[3];
  float* out = (float*)d_out;

  // Workspace layout (all 256B aligned); total ~8.6 MB.
  char* ws = (char*)d_ws;
  bf16*  xh      = (bf16*) (ws + 0);          // 8192*128*2 = 2 MB
  bf16*  xl      = (bf16*) (ws + 2097152);
  bf16*  yh      = (bf16*) (ws + 4194304);
  bf16*  yl      = (bf16*) (ws + 6291456);
  float* xsq     = (float*)(ws + 8388608);    // 32 KB
  float* ysq     = (float*)(ws + 8421376);    // 32 KB
  float* picked  = (float*)(ws + 8454144);    // 32 KB
  float* partial = (float*)(ws + 8486912);    // 4*32 KB

  // 16384 rows, 1 wave/row, 8 waves/block.
  CLoss_prep_kernel<<<2048, 256, 0, stream>>>(feat, feat2, xh, xl, yh, yl, xsq, ysq);
  // 8192 rows, 1 wave/row.
  CLoss_picked_kernel<<<1024, 256, 0, stream>>>(feat, feat2, labels, temp, picked);
  // 512 row-tiles x 4 column chunks = 2048 waves = 256 blocks.
  CLoss_main_kernel<<<256, 256, 0, stream>>>(xh, xl, yh, yl, xsq, ysq, temp, partial);
  // Deterministic scalar reduction.
  CLoss_reduce_kernel<<<1, 256, 0, stream>>>(partial, picked, out);
}
